// InteractionBlock_14370960572978
// MI455X (gfx1250) — matrix operand order repaired and checked by reading, not compiled, not run on hardware
//
#include <hip/hip_runtime.h>
#include <hip/hip_bf16.h>

#define N_NODESC 50000     /* % 16 == 0 -> every 16-row wave tile is full */
#define N_EDGESC 800000    /* % 64 == 0 -> every 64-edge block tile is full */
#define HID 128
#define NG 50
#define SHIFT_LN2 0.69314718055994531f
#define KP 136   /* 128 + 8 pad (bf16 elems): conflict-free LDS B-frag reads */
#define KP1 72   /* 64 + 8 pad for the Gaussian layer (K padded 50->64)     */
#define NTILES_E (N_EDGESC / 64)   /* 12500 */
#define EDGE_BLOCKS 1250           /* 10 tiles per persistent block */

typedef __attribute__((ext_vector_type(16))) __bf16 v16bf;
typedef __attribute__((ext_vector_type(8)))  float  v8f;

__device__ __forceinline__ v8f wmma_bf16(v16bf a, v16bf b, v8f c) {
  // D = A(16x32) * B(32x16) + C, fp32 accumulate
  return __builtin_amdgcn_wmma_f32_16x16x32_bf16(
      /*neg_a=*/false, a, /*neg_b=*/false, b,
      /*c_mod=*/(short)0, c, /*reuse_a=*/false, /*reuse_b=*/false);
}

__device__ __forceinline__ float ssp(float v) {
  // shifted softplus: softplus(v) - ln2, numerically stable
  return fmaxf(v, 0.0f) + log1pf(__expf(-fabsf(v))) - SHIFT_LN2;
}

// ---- WMMA fragment loaders (wave32 layouts per CDNA5 ISA 7.12.2) ----------
// A 16x32 bf16: lane holds row m = lane&15; half-waves split K 0-7/16-23 vs 8-15/24-31.
__device__ __forceinline__ v16bf afrag_f32(const float* rowp, int k0, int kmax) {
  int lane = threadIdx.x & 31;
  int kb = k0 + ((lane & 16) ? 8 : 0);
  v16bf a;
#pragma unroll
  for (int j = 0; j < 16; ++j) {
    int k = kb + (j < 8 ? j : j + 8);
    a[j] = (__bf16)((k < kmax) ? rowp[k] : 0.0f);
  }
  return a;
}

__device__ __forceinline__ v16bf afrag_bf16(const __bf16* rowp, int k0) {
  int lane = threadIdx.x & 31;
  int kb = k0 + ((lane & 16) ? 8 : 0);
  v16bf a;
#pragma unroll
  for (int j = 0; j < 16; ++j)
    a[j] = rowp[kb + (j < 8 ? j : j + 8)];
  return a;
}

// B 32x16 bf16 from LDS weights stored [out][kpad] row-major (W^T access):
// lane holds column n = n0 + (lane&15); half-waves split K 0-7/16-23 vs 8-15/24-31.
__device__ __forceinline__ v16bf bfrag_lds(const __bf16* w, int n0, int k0, int kpad) {
  int lane = threadIdx.x & 31;
  const __bf16* p = w + (n0 + (lane & 15)) * kpad + k0 + ((lane & 16) ? 8 : 0);
  v16bf b;
#pragma unroll
  for (int j = 0; j < 16; ++j) b[j] = p[j < 8 ? j : j + 8];
  return b;
}

// ---- Kernel 1: h = bf16(x @ lin1_w^T)  ([N,128] x [128,128]) --------------
__global__ __launch_bounds__(256) void proj_kernel(const float* __restrict__ x,
                                                   const float* __restrict__ lin1_w,
                                                   __bf16* __restrict__ hp) {
  __shared__ __bf16 lw[HID * KP];
  int tid = threadIdx.x;
  for (int idx = tid; idx < HID * HID; idx += 256) {
    int f = idx >> 7, k = idx & 127;
    lw[f * KP + k] = (__bf16)lin1_w[idx];
  }
  __syncthreads();

  int wv = tid >> 5, lane = tid & 31, m = lane & 15;
  int row0 = blockIdx.x * 128 + wv * 16;
  if (row0 >= N_NODESC) return;               // N%16==0: tile is full or absent

  const float* rowp = x + (size_t)(row0 + m) * HID;
  v16bf a[4];
#pragma unroll
  for (int ks = 0; ks < 4; ++ks) a[ks] = afrag_f32(rowp, ks * 32, HID);

  int rbase = row0 + ((lane & 16) ? 8 : 0);
#pragma unroll
  for (int nt = 0; nt < 8; ++nt) {
    v8f acc = {};
#pragma unroll
    for (int ks = 0; ks < 4; ++ks)
      acc = wmma_bf16(a[ks], bfrag_lds(lw, nt * 16, ks * 32, KP), acc);
    int n = nt * 16 + m;
#pragma unroll
    for (int i = 0; i < 8; ++i)
      hp[(size_t)(rbase + i) * HID + n] = (__bf16)acc[i];
  }
}

// ---- Kernel 2: edge MLP + cutoff + gather-multiply-scatter ----------------
// Persistent blocks: 4 waves, grid-stride over 64-edge tiles. w2 resident in
// LDS for block lifetime; w1 region doubles as activation staging per tile.
__global__ __launch_bounds__(128) void edge_kernel(
    const float* __restrict__ edge_attr, const float* __restrict__ edge_weight,
    const long long* __restrict__ edge_index,
    const float* __restrict__ mlp_w1, const float* __restrict__ mlp_b1,
    const float* __restrict__ mlp_w2, const float* __restrict__ mlp_b2,
    const __bf16* __restrict__ hp, float* __restrict__ agg) {
  __shared__ __bf16 r0[HID * KP1];        // 9216 elems >= staging 4*16*136 = 8704
  __shared__ __bf16 w2l[HID * KP];
  __shared__ float lb1[HID], lb2[HID];
  __shared__ float lC[4][16];
  __shared__ int lsrc[4][16], ldst[4][16];

  int tid = threadIdx.x, wv = tid >> 5, lane = tid & 31, m = lane & 15;

  for (int idx = tid; idx < HID * HID; idx += 128) {
    int f = idx >> 7, k = idx & 127;
    w2l[f * KP + k] = (__bf16)mlp_w2[idx];
  }
  if (tid < HID) { lb1[tid] = mlp_b1[tid]; lb2[tid] = mlp_b2[tid]; }

  for (int tile = blockIdx.x; tile < NTILES_E; tile += gridDim.x) {
    // (re)load w1 into region0 (K padded 50 -> 64 with zeros)
    for (int idx = tid; idx < HID * 64; idx += 128) {
      int f = idx >> 6, k = idx & 63;
      r0[f * KP1 + k] = (__bf16)((k < NG) ? mlp_w1[f * NG + k] : 0.0f);
    }
    int e0 = tile * 64 + wv * 16;           // always fully in range (E%64==0)
    if (lane < 16) {
      int e = e0 + lane;
      lsrc[wv][lane] = (int)edge_index[e];
      ldst[wv][lane] = (int)edge_index[N_EDGESC + e];
      lC[wv][lane] = 0.5f * (__cosf(edge_weight[e] * 0.31415926535897932f) + 1.0f);
    }
    __syncthreads();

    // Layer 1: [16,64] x [64,128] + bias, shifted-softplus
    const float* erow = edge_attr + (size_t)(e0 + m) * NG;
    v16bf a0 = afrag_f32(erow, 0, NG);
    v16bf a1 = afrag_f32(erow, 32, NG);
    float act[8][8];
#pragma unroll
    for (int nt = 0; nt < 8; ++nt) {
      v8f acc = {};
      acc = wmma_bf16(a0, bfrag_lds(r0, nt * 16, 0, KP1), acc);
      acc = wmma_bf16(a1, bfrag_lds(r0, nt * 16, 32, KP1), acc);
      float bias = lb1[nt * 16 + m];
#pragma unroll
      for (int i = 0; i < 8; ++i) act[nt][i] = ssp(acc[i] + bias);
    }
    __syncthreads();  // all waves done reading w1 -> reuse region0 as staging

    __bf16* stag = r0 + wv * (16 * KP);
#pragma unroll
    for (int nt = 0; nt < 8; ++nt) {
      int n = nt * 16 + m;
#pragma unroll
      for (int i = 0; i < 8; ++i) {
        int mi = ((lane & 16) ? 8 : 0) + i;
        stag[mi * KP + n] = (__bf16)act[nt][i];
      }
    }
    // same-wave DS RAW handled by compiler-inserted s_wait_dscnt
    const __bf16* srow = stag + m * KP;
    v16bf a2[4];
#pragma unroll
    for (int ks = 0; ks < 4; ++ks) a2[ks] = afrag_bf16(srow, ks * 32);
    __syncthreads();  // staging consumed -> region0 free for next tile's w1

    // Layer 2: [16,128] x [128,128] + bias, * C, * h[src], scatter to agg[dst]
#pragma unroll
    for (int nt = 0; nt < 8; ++nt) {
      v8f acc = {};
#pragma unroll
      for (int ks = 0; ks < 4; ++ks)
        acc = wmma_bf16(a2[ks], bfrag_lds(w2l, nt * 16, ks * 32, KP), acc);
      int n = nt * 16 + m;
      float bias = lb2[n];
#pragma unroll
      for (int i = 0; i < 8; ++i) {
        int mi = ((lane & 16) ? 8 : 0) + i;
        float wval = (acc[i] + bias) * lC[wv][mi];
        float hval = (float)hp[(size_t)lsrc[wv][mi] * HID + n];
        unsafeAtomicAdd(&agg[(size_t)ldst[wv][mi] * HID + n], wval * hval);
      }
    }
  }
}

// ---- Kernel 3a: t = bf16(ssp(agg @ lin2_w^T + b)) -------------------------
__global__ __launch_bounds__(256) void node_out1_kernel(const float* __restrict__ agg,
                                                        const float* __restrict__ lin2_w,
                                                        const float* __restrict__ lin2_b,
                                                        __bf16* __restrict__ t) {
  __shared__ __bf16 lw[HID * KP];
  __shared__ float lb[HID];
  int tid = threadIdx.x;
  for (int idx = tid; idx < HID * HID; idx += 256) {
    int f = idx >> 7, k = idx & 127;
    lw[f * KP + k] = (__bf16)lin2_w[idx];
  }
  if (tid < HID) lb[tid] = lin2_b[tid];
  __syncthreads();

  int wv = tid >> 5, lane = tid & 31, m = lane & 15;
  int row0 = blockIdx.x * 128 + wv * 16;
  if (row0 >= N_NODESC) return;

  const float* rowp = agg + (size_t)(row0 + m) * HID;
  v16bf a[4];
#pragma unroll
  for (int ks = 0; ks < 4; ++ks) a[ks] = afrag_f32(rowp, ks * 32, HID);

  int rbase = row0 + ((lane & 16) ? 8 : 0);
#pragma unroll
  for (int nt = 0; nt < 8; ++nt) {
    v8f acc = {};
#pragma unroll
    for (int ks = 0; ks < 4; ++ks)
      acc = wmma_bf16(a[ks], bfrag_lds(lw, nt * 16, ks * 32, KP), acc);
    int n = nt * 16 + m;
    float bias = lb[n];
#pragma unroll
    for (int i = 0; i < 8; ++i)
      t[(size_t)(rbase + i) * HID + n] = (__bf16)ssp(acc[i] + bias);
  }
}

// ---- Kernel 3b: out = t @ lin_w^T + b -------------------------------------
__global__ __launch_bounds__(256) void node_out2_kernel(const __bf16* __restrict__ t,
                                                        const float* __restrict__ lin_w,
                                                        const float* __restrict__ lin_b,
                                                        float* __restrict__ out) {
  __shared__ __bf16 lw[HID * KP];
  __shared__ float lb[HID];
  int tid = threadIdx.x;
  for (int idx = tid; idx < HID * HID; idx += 256) {
    int f = idx >> 7, k = idx & 127;
    lw[f * KP + k] = (__bf16)lin_w[idx];
  }
  if (tid < HID) lb[tid] = lin_b[tid];
  __syncthreads();

  int wv = tid >> 5, lane = tid & 31, m = lane & 15;
  int row0 = blockIdx.x * 128 + wv * 16;
  if (row0 >= N_NODESC) return;

  const __bf16* rowp = t + (size_t)(row0 + m) * HID;
  v16bf a[4];
#pragma unroll
  for (int ks = 0; ks < 4; ++ks) a[ks] = afrag_bf16(rowp, ks * 32);

  int rbase = row0 + ((lane & 16) ? 8 : 0);
#pragma unroll
  for (int nt = 0; nt < 8; ++nt) {
    v8f acc = {};
#pragma unroll
    for (int ks = 0; ks < 4; ++ks)
      acc = wmma_bf16(a[ks], bfrag_lds(lw, nt * 16, ks * 32, KP), acc);
    int n = nt * 16 + m;
    float bias = lb[n];
#pragma unroll
    for (int i = 0; i < 8; ++i)
      out[(size_t)(rbase + i) * HID + n] = acc[i] + bias;
  }
}

extern "C" void kernel_launch(void* const* d_in, const int* in_sizes, int n_in,
                              void* d_out, int out_size, void* d_ws, size_t ws_size,
                              hipStream_t stream) {
  const float*     x   = (const float*)d_in[0];
  const long long* ei  = (const long long*)d_in[1];   // int64 per reference
  const float*     ew  = (const float*)d_in[2];
  const float*     ea  = (const float*)d_in[3];
  const float*     w1  = (const float*)d_in[4];
  const float*     b1  = (const float*)d_in[5];
  const float*     w2  = (const float*)d_in[6];
  const float*     b2  = (const float*)d_in[7];
  const float*     l1w = (const float*)d_in[8];
  const float*     l2w = (const float*)d_in[9];
  const float*     l2b = (const float*)d_in[10];
  const float*     lnw = (const float*)d_in[11];
  const float*     lnb = (const float*)d_in[12];
  float* out = (float*)d_out;

  __bf16* hpb = (__bf16*)d_ws;                              // [N,128] bf16
  float*  agg = (float*)(hpb + (size_t)N_NODESC * HID);     // [N,128] f32
  __bf16* t   = (__bf16*)(agg + (size_t)N_NODESC * HID);    // [N,128] bf16

  hipMemsetAsync(agg, 0, (size_t)N_NODESC * HID * sizeof(float), stream);

  int nblk = (N_NODESC + 127) / 128;
  proj_kernel<<<nblk, 256, 0, stream>>>(x, l1w, hpb);
  edge_kernel<<<EDGE_BLOCKS, 128, 0, stream>>>(ea, ew, ei, w1, b1, w2, b2, hpb, agg);
  node_out1_kernel<<<nblk, 256, 0, stream>>>(agg, l2w, l2b, t);
  node_out2_kernel<<<nblk, 256, 0, stream>>>(t, lnw, lnb, out);
}
